// RnnGnn_89172110999587
// MI455X (gfx1250) — matrix operand order repaired and checked by reading, not compiled, not run on hardware
//
#include <hip/hip_runtime.h>
#include <hip/hip_bf16.h>

typedef __attribute__((ext_vector_type(16))) __bf16 bf16x16;
typedef __attribute__((ext_vector_type(8)))  __bf16 bf16x8;
typedef __attribute__((ext_vector_type(4)))  __bf16 bf16x4;
typedef __attribute__((ext_vector_type(8)))  float  f32x8;
typedef __attribute__((ext_vector_type(4)))  float  f32x4;
typedef __attribute__((ext_vector_type(4)))  int    i32x4;

#define T_STEPS    2048
#define NN         100
#define F_IN       128
#define H_DIM      512
#define G3H        1536
#define ROWS_TOTAL (T_STEPS * NN)   // 204800
#define M_PAD      112
#define KNN_K      8
#define EMB_DIM    128
#define FLAT_IN    32
#define FLAT_OUT   64
#define GIN        704
#define GH         256
#define GOUT       128
#define GRU_WGS    16
#define LDK        40   // transposed-B LDS row stride (80B; 20-dword lane stride -> 16 distinct banks)

#if __has_builtin(__builtin_amdgcn_global_load_async_to_lds_b128)
#define HAVE_ASYNC_LDS 1
#else
#define HAVE_ASYNC_LDS 0
#endif

// Builtin signature (from hipcc diagnostic): (v4i AS1*, v4i AS3*, Ii offset, Ii cpol)
#define AS1V4(p) ((__attribute__((address_space(1))) i32x4*)(unsigned long long)(p))
#define AS3V4(p) ((__attribute__((address_space(3))) i32x4*)(unsigned long long)(p))

// ---------------- WMMA fragment helpers (CDNA5 documented layouts) ----------

// 16-bit A 16x32: lane<16 -> row, half selects K {0..7,16..23} / {8..15,24..31}.
__device__ __forceinline__ bf16x16 load_a_frag(const __bf16* As, int lda) {
  int lane = threadIdx.x & 31;
  int row  = lane & 15;
  int half = lane >> 4;
  const bf16x8* p = (const bf16x8*)(As + row * lda + half * 8);
  bf16x8 lo = p[0];        // K = half*8 .. +7        (ds_load_b128)
  bf16x8 hi = p[2];        // K = 16 + half*8 .. +7   (ds_load_b128)
  return __builtin_shufflevector(lo, hi, 0, 1, 2, 3, 4, 5, 6, 7,
                                 8, 9, 10, 11, 12, 13, 14, 15);
}

// 16-bit B 32x16: lane (half, n) needs K {half*8..+7, 16+half*8..+7} at fixed column n.
// BsT stored transposed [n][k] with stride LDK -> two 16-byte chunks per lane.
__device__ __forceinline__ bf16x16 load_b_frag(const __bf16* BsT) {
  int lane = threadIdx.x & 31;
  int col  = lane & 15;
  int half = lane >> 4;
  const bf16x8* p = (const bf16x8*)(BsT + col * LDK + half * 8);
  bf16x8 lo = p[0];        // ds_load_b128
  bf16x8 hi = p[2];        // ds_load_b128
  return __builtin_shufflevector(lo, hi, 0, 1, 2, 3, 4, 5, 6, 7,
                                 8, 9, 10, 11, 12, 13, 14, 15);
}

__device__ __forceinline__ f32x8 wmma_bf16(bf16x16 a, bf16x16 b, f32x8 c) {
  return __builtin_amdgcn_wmma_f32_16x16x32_bf16(false, a, false, b, (short)0, c,
                                                 false, false);
}

// ---------------- prep: blocked bf16 weights, zero buffers, flat linear -----
// Wb_ih: [cb=24][kc=4][c=64][k=32]  (B-tile image for k_xw_gemm)
// Wb_hh: [wg=16][kc=16][c=96][k=32] (B-tile image for k_gru, gate-mapped cols)

__global__ void k_prep(const float* __restrict__ Wih, const float* __restrict__ Whh,
                       const float* __restrict__ flat, const float* __restrict__ flat_W,
                       const float* __restrict__ flat_b,
                       __bf16* __restrict__ Wb_ih, __bf16* __restrict__ Wb_hh,
                       __bf16* __restrict__ hbuf, float* __restrict__ flat_e,
                       float* __restrict__ agg1, float* __restrict__ agg2,
                       unsigned* __restrict__ cnt, unsigned* __restrict__ bar) {
  int idx0 = blockIdx.x * blockDim.x + threadIdx.x;
  int stride = gridDim.x * blockDim.x;
  for (int i = idx0; i < G3H * F_IN; i += stride) {
    int k  = i & 31;
    int c  = (i >> 5) & 63;
    int kc = (i >> 11) & 3;
    int cb = i >> 13;
    Wb_ih[i] = (__bf16)Wih[(size_t)(cb * 64 + c) * F_IN + kc * 32 + k];
  }
  for (int i = idx0; i < GRU_WGS * 16 * 96 * 32; i += stride) {
    int k  = i & 31;
    int c  = (i >> 5) % 96;
    int kc = (i / (96 * 32)) & 15;
    int wg = i / (16 * 96 * 32);
    int g  = (c >> 5) * H_DIM + wg * 32 + (c & 31);
    Wb_hh[i] = (__bf16)Whh[(size_t)g * H_DIM + kc * 32 + k];
  }
  for (int i = idx0; i < 2 * M_PAD * H_DIM; i += stride) hbuf[i] = (__bf16)0.0f;
  for (int i = idx0; i < NN * FLAT_OUT; i += stride) {
    int n = i / FLAT_OUT, o = i % FLAT_OUT;
    float s = flat_b[o];
    for (int c = 0; c < FLAT_IN; ++c) s += flat[n * FLAT_IN + c] * flat_W[o * FLAT_IN + c];
    flat_e[i] = s;
  }
  for (int i = idx0; i < NN * GIN; i += stride) agg1[i] = 0.0f;
  for (int i = idx0; i < NN * GH;  i += stride) agg2[i] = 0.0f;
  for (int i = idx0; i < NN;       i += stride) cnt[i] = 0u;
  if (idx0 == 0) *bar = 0u;
}

// ---------------- xW = x @ Wih^T + bih  (bf16 WMMA GEMM, 80.5 GFLOP) --------

__global__ __launch_bounds__(256) void k_xw_gemm(const float* __restrict__ X,
                                                 const __bf16* __restrict__ Wb_ih,
                                                 const float* __restrict__ bih,
                                                 __bf16* __restrict__ xW) {
  __shared__ __bf16 As[128 * 40];   // 128x32 bf16 (+pad)
  __shared__ __bf16 BsT[64 * LDK];  // 64 cols x 32 k, fragment-transposed
  const int tid  = threadIdx.x;
  const int wave = tid >> 5;
  const int r0 = blockIdx.x * 128;
  const int c0 = blockIdx.y * 64;

  f32x8 acc[4];
#pragma unroll
  for (int i = 0; i < 4; ++i) acc[i] = (f32x8){0, 0, 0, 0, 0, 0, 0, 0};

  for (int kk = 0; kk < F_IN; kk += 32) {
    // A tile: vectorized f32 load + bf16 pack, 4 elems/thread/iter
    for (int i = tid; i < 128 * 8; i += 256) {
      int r = i >> 3, q = i & 7;
      int row = r0 + r;
      f32x4 v = (f32x4){0, 0, 0, 0};
      if (row < ROWS_TOTAL) v = *(const f32x4*)(X + (size_t)row * F_IN + kk + q * 4);
      bf16x4 b4;
#pragma unroll
      for (int u = 0; u < 4; ++u) b4[u] = (__bf16)v[u];
      *(bf16x4*)(As + r * 40 + q * 4) = b4;
    }
    // B tile: contiguous copy from blocked image, 16B chunks
    const __bf16* wsrc = Wb_ih + ((size_t)(blockIdx.y * 4 + (kk >> 5)) * 64) * 32;
    for (int i = tid; i < 64 * 4; i += 256) {
      int c = i >> 2, part = i & 3;
      *(bf16x8*)(BsT + c * LDK + part * 8) = *(const bf16x8*)(wsrc + c * 32 + part * 8);
    }
    __syncthreads();
    bf16x16 a = load_a_frag(As + wave * 16 * 40, 40);
#pragma unroll
    for (int nt = 0; nt < 4; ++nt) {
      bf16x16 b = load_b_frag(BsT + nt * 16 * LDK);
      acc[nt] = wmma_bf16(a, b, acc[nt]);
    }
    __syncthreads();
  }
  int lane = tid & 31;
  int n0 = lane & 15, half = lane >> 4;
#pragma unroll
  for (int nt = 0; nt < 4; ++nt)
#pragma unroll
    for (int i = 0; i < 8; ++i) {
      int row = r0 + wave * 16 + half * 8 + i;
      int col = c0 + nt * 16 + n0;
      if (row < ROWS_TOTAL)
        xW[(size_t)row * G3H + col] = (__bf16)(acc[nt][i] + bih[col]);
    }
}

// ---------------- GRU recurrence: persistent, 16 WGs, column-split ----------

__global__ __launch_bounds__(256) void k_gru(const __bf16* __restrict__ xW,
                                             const __bf16* __restrict__ Wb_hh,
                                             const float* __restrict__ bhh,
                                             __bf16* __restrict__ hbuf,  // [2][112][512]
                                             float* __restrict__ enc_out,
                                             unsigned* __restrict__ bar) {
  __shared__ __bf16 Hs[M_PAD * 40];   // 112x32 bf16 (+pad)
  __shared__ __bf16 BsT[96 * LDK];    // 96 gate-cols x 32 k, fragment-transposed
  __shared__ float  HW[M_PAD * 96];   // f32 gate pre-activations
  const int tid  = threadIdx.x;
  const int wave = tid >> 5;
  const int j0   = blockIdx.x * 32;   // this WG's H-column slice
  const __bf16* wbase = Wb_hh + (size_t)blockIdx.x * 16 * 96 * 32;

  for (int t = 0; t < T_STEPS; ++t) {
    const __bf16* hcur = hbuf + (size_t)(t & 1) * M_PAD * H_DIM;
    __bf16*       hnxt = hbuf + (size_t)((t + 1) & 1) * M_PAD * H_DIM;

    f32x8 acc[6];
#pragma unroll
    for (int i = 0; i < 6; ++i) acc[i] = (f32x8){0, 0, 0, 0, 0, 0, 0, 0};

    for (int kk = 0; kk < H_DIM; kk += 32) {
      // h tile: vectorized 16B copies
      for (int i = tid; i < M_PAD * 4; i += 256) {
        int r = i >> 2, part = i & 3;
        *(bf16x8*)(Hs + r * 40 + part * 8) =
            *(const bf16x8*)(hcur + r * H_DIM + kk + part * 8);
      }
      // Whh tile: contiguous from blocked image; async copy to LDS when available
      const __bf16* wsrc = wbase + (size_t)(kk >> 5) * 96 * 32;
#if HAVE_ASYNC_LDS
      for (int i = tid; i < 96 * 4; i += 256) {
        int c = i >> 2, part = i & 3;
        __builtin_amdgcn_global_load_async_to_lds_b128(
            AS1V4(wsrc + c * 32 + part * 8),
            AS3V4(BsT + c * LDK + part * 8), 0, 0);
      }
#else
      for (int i = tid; i < 96 * 4; i += 256) {
        int c = i >> 2, part = i & 3;
        *(bf16x8*)(BsT + c * LDK + part * 8) = *(const bf16x8*)(wsrc + c * 32 + part * 8);
      }
#endif
      if (kk + 32 < H_DIM && tid < 48)  // L2 streaming hint for next 6KB block
        __builtin_prefetch(wbase + (size_t)((kk >> 5) + 1) * 96 * 32 + tid * 128, 0, 0);
#if HAVE_ASYNC_LDS
      asm volatile("s_wait_asynccnt 0x0" ::: "memory");
#endif
      __syncthreads();
      if (wave < 7) {  // waves 0..6 own M tiles 0..6 (rows 0..111)
        bf16x16 a = load_a_frag(Hs + wave * 16 * 40, 40);
#pragma unroll
        for (int nt = 0; nt < 6; ++nt) {
          bf16x16 b = load_b_frag(BsT + nt * 16 * LDK);
          acc[nt] = wmma_bf16(a, b, acc[nt]);
        }
      }
      __syncthreads();
    }
    if (wave < 7) {
      int lane = tid & 31;
      int n0 = lane & 15, half = lane >> 4;
#pragma unroll
      for (int nt = 0; nt < 6; ++nt)
#pragma unroll
        for (int i = 0; i < 8; ++i)
          HW[(wave * 16 + half * 8 + i) * 96 + nt * 16 + n0] = acc[nt][i];
    }
    __syncthreads();

    // gate math, 8-wide vectorized
    const __bf16* xrow = xW + (size_t)t * NN * G3H;
    for (int i = tid; i < NN * 4; i += 256) {
      int n = i >> 2, c8 = (i & 3) * 8;
      int j = j0 + c8;
      bf16x8 xr8 = *(const bf16x8*)(xrow + (size_t)n * G3H + j);
      bf16x8 xz8 = *(const bf16x8*)(xrow + (size_t)n * G3H + H_DIM + j);
      bf16x8 xn8 = *(const bf16x8*)(xrow + (size_t)n * G3H + 2 * H_DIM + j);
      bf16x8 hp8 = *(const bf16x8*)(hcur + n * H_DIM + j);
      bf16x8 ho8;
#pragma unroll
      for (int u = 0; u < 8; ++u) {
        float hr = HW[n * 96 + c8 + u]      + bhh[j + u];
        float hz = HW[n * 96 + 32 + c8 + u] + bhh[H_DIM + j + u];
        float hn = HW[n * 96 + 64 + c8 + u] + bhh[2 * H_DIM + j + u];
        float r  = 1.0f / (1.0f + expf(-((float)xr8[u] + hr)));
        float z  = 1.0f / (1.0f + expf(-((float)xz8[u] + hz)));
        float ng = tanhf((float)xn8[u] + r * hn);
        float hh = (1.0f - z) * ng + z * (float)hp8[u];
        ho8[u] = (__bf16)hh;
        if (t == T_STEPS - 1) enc_out[n * H_DIM + j + u] = hh;
      }
      *(bf16x8*)(hnxt + n * H_DIM + j) = ho8;
    }
    __syncthreads();
    if (tid == 0) {  // global generation barrier across 16 co-resident WGs
      __threadfence();
      atomicAdd(bar, 1u);
      unsigned target = (unsigned)(t + 1) * GRU_WGS;
      while (atomicAdd(bar, 0u) < target) __builtin_amdgcn_s_sleep(1);
      __threadfence();
    }
    __syncthreads();
  }
}

// ---------------- kNN (cosine top-8) ----------------------------------------

__global__ __launch_bounds__(128) void k_knn(const float* __restrict__ emb,
                                             int* __restrict__ edges,
                                             unsigned* __restrict__ cnt) {
  __shared__ float A[NN * EMB_DIM];
  int tid = threadIdx.x;
  for (int n = tid; n < NN; n += 128) {
    float s = 0.0f;
    for (int c = 0; c < EMB_DIM; ++c) { float v = emb[n * EMB_DIM + c]; s += v * v; }
    float inv = 1.0f / fmaxf(sqrtf(s), 1e-8f);
    for (int c = 0; c < EMB_DIM; ++c) A[n * EMB_DIM + c] = emb[n * EMB_DIM + c] * inv;
  }
  __syncthreads();
  for (int i = tid; i < NN; i += 128) {
    float bv[KNN_K]; int bi[KNN_K];
    for (int k = 0; k < KNN_K; ++k) { bv[k] = -1e30f; bi[k] = 0; }
    for (int j = 0; j < NN; ++j) {
      float d = 0.0f;
      if (j != i)
        for (int c = 0; c < EMB_DIM; ++c) d += A[i * EMB_DIM + c] * A[j * EMB_DIM + c];
      int mink = 0; float minv = bv[0];
      for (int k = 1; k < KNN_K; ++k) if (bv[k] < minv) { minv = bv[k]; mink = k; }
      if (d > minv) { bv[mink] = d; bi[mink] = j; }
    }
    for (int k = 0; k < KNN_K; ++k) {
      edges[i * KNN_K + k] = bi[k];
      atomicAdd(&cnt[bi[k]], 1u);
    }
  }
}

// ---------------- GNN small kernels -----------------------------------------

__global__ void k_build_gnn(const float* __restrict__ enc, const float* __restrict__ fe,
                            const float* __restrict__ emb, float* __restrict__ gnn_in) {
  int idx = blockIdx.x * blockDim.x + threadIdx.x;
  if (idx >= NN * GIN) return;
  int n = idx / GIN, c = idx % GIN;
  float v;
  if (c < H_DIM)                 v = enc[n * H_DIM + c];
  else if (c < H_DIM + FLAT_OUT) v = fe[n * FLAT_OUT + (c - H_DIM)];
  else                           v = emb[n * EMB_DIM + (c - H_DIM - FLAT_OUT)];
  gnn_in[idx] = v;
}

__global__ void k_agg(const float* __restrict__ x, const int* __restrict__ edges,
                      float* __restrict__ agg, int dim) {
  int idx = blockIdx.x * blockDim.x + threadIdx.x;
  if (idx >= NN * KNN_K * dim) return;
  int e = idx / dim, c = idx % dim;
  int srcn = e / KNN_K;
  int tg = edges[e];
  atomicAdd(&agg[(size_t)tg * dim + c], x[(size_t)srcn * dim + c]);
}

__global__ void k_sage(const float* __restrict__ agg, const float* __restrict__ x,
                       const unsigned* __restrict__ cnt, const float* __restrict__ Wl,
                       const float* __restrict__ Wr, const float* __restrict__ b,
                       float* __restrict__ out, int din, int dout, int do_relu) {
  int idx = blockIdx.x * blockDim.x + threadIdx.x;
  if (idx >= NN * dout) return;
  int n = idx / dout, o = idx % dout;
  float inv = 1.0f / fmaxf((float)cnt[n], 1.0f);
  float s = b[o];
  for (int c = 0; c < din; ++c)
    s += agg[(size_t)n * din + c] * inv * Wl[(size_t)o * din + c] +
         x[(size_t)n * din + c] * Wr[(size_t)o * din + c];
  out[idx] = do_relu ? fmaxf(s, 0.0f) : s;
}

__global__ void k_out(const float* __restrict__ g2, const float* __restrict__ enc,
                      const float* __restrict__ out_W, const float* __restrict__ out_b,
                      float* __restrict__ out) {
  int n = blockIdx.x * blockDim.x + threadIdx.x;
  if (n >= NN) return;
  float s = out_b[0];
  for (int c = 0; c < GOUT; ++c)  s += g2[n * GOUT + c] * out_W[c];
  for (int c = 0; c < H_DIM; ++c) s += enc[n * H_DIM + c] * out_W[GOUT + c];
  out[n] = 1.0f / (1.0f + expf(-s));
}

// ---------------- launch ----------------------------------------------------

extern "C" void kernel_launch(void* const* d_in, const int* in_sizes, int n_in,
                              void* d_out, int out_size, void* d_ws, size_t ws_size,
                              hipStream_t stream) {
  const float* node_feat = (const float*)d_in[0];
  const float* flat      = (const float*)d_in[1];
  const float* emb       = (const float*)d_in[2];
  const float* gru_Wih   = (const float*)d_in[3];
  const float* gru_Whh   = (const float*)d_in[4];
  const float* gru_bih   = (const float*)d_in[5];
  const float* gru_bhh   = (const float*)d_in[6];
  const float* flat_W    = (const float*)d_in[7];
  const float* flat_b    = (const float*)d_in[8];
  const float* s1_Wl     = (const float*)d_in[9];
  const float* s1_Wr     = (const float*)d_in[10];
  const float* s1_b      = (const float*)d_in[11];
  const float* s2_Wl     = (const float*)d_in[12];
  const float* s2_Wr     = (const float*)d_in[13];
  const float* s2_b      = (const float*)d_in[14];
  const float* out_W     = (const float*)d_in[15];
  const float* out_b     = (const float*)d_in[16];
  float* out = (float*)d_out;

  char* ws = (char*)d_ws;
  size_t off = 0;
  auto alloc = [&](size_t bytes) -> char* {
    off = (off + 255) & ~(size_t)255;
    char* p = ws + off;
    off += bytes;
    return p;
  };
  __bf16* xW_bf   = (__bf16*)alloc((size_t)ROWS_TOTAL * G3H * 2);  // 629 MB
  __bf16* Wb_ih   = (__bf16*)alloc((size_t)F_IN * G3H * 2);
  __bf16* Wb_hh   = (__bf16*)alloc((size_t)H_DIM * G3H * 2);
  __bf16* hbuf    = (__bf16*)alloc((size_t)2 * M_PAD * H_DIM * 2);
  float* enc_out  = (float*)alloc((size_t)NN * H_DIM * 4);
  float* flat_e   = (float*)alloc((size_t)NN * FLAT_OUT * 4);
  float* gnn_in   = (float*)alloc((size_t)NN * GIN * 4);
  float* agg1     = (float*)alloc((size_t)NN * GIN * 4);
  float* h1       = (float*)alloc((size_t)NN * GH * 4);
  float* agg2     = (float*)alloc((size_t)NN * GH * 4);
  float* g2       = (float*)alloc((size_t)NN * GOUT * 4);
  int* edges      = (int*)alloc((size_t)NN * KNN_K * 4);
  unsigned* cnt   = (unsigned*)alloc((size_t)NN * 4);
  unsigned* bar   = (unsigned*)alloc(256);

  k_prep<<<1024, 256, 0, stream>>>(gru_Wih, gru_Whh, flat, flat_W, flat_b,
                                   Wb_ih, Wb_hh, hbuf, flat_e, agg1, agg2, cnt, bar);

  dim3 g_xw(ROWS_TOTAL / 128, G3H / 64);
  k_xw_gemm<<<g_xw, 256, 0, stream>>>(node_feat, Wb_ih, gru_bih, xW_bf);

  k_knn<<<1, 128, 0, stream>>>(emb, edges, cnt);

  k_gru<<<GRU_WGS, 256, 0, stream>>>(xW_bf, Wb_hh, gru_bhh, hbuf, enc_out, bar);

  k_build_gnn<<<(NN * GIN + 255) / 256, 256, 0, stream>>>(enc_out, flat_e, emb, gnn_in);
  k_agg<<<(NN * KNN_K * GIN + 255) / 256, 256, 0, stream>>>(gnn_in, edges, agg1, GIN);
  k_sage<<<(NN * GH + 255) / 256, 256, 0, stream>>>(agg1, gnn_in, cnt, s1_Wl, s1_Wr,
                                                    s1_b, h1, GIN, GH, 1);
  k_agg<<<(NN * KNN_K * GH + 255) / 256, 256, 0, stream>>>(h1, edges, agg2, GH);
  k_sage<<<(NN * GOUT + 255) / 256, 256, 0, stream>>>(agg2, h1, cnt, s2_Wl, s2_Wr,
                                                      s2_b, g2, GH, GOUT, 0);
  k_out<<<1, 128, 0, stream>>>(g2, enc_out, out_W, out_b, out);

  (void)in_sizes; (void)n_in; (void)out_size; (void)ws_size;
}
